// MultiHeadAttention_1408749273781
// MI455X (gfx1250) — matrix operand order repaired and checked by reading, not compiled, not run on hardware
//
#include <hip/hip_runtime.h>

typedef _Float16 f16;
typedef __attribute__((ext_vector_type(16))) _Float16 v16h;
typedef __attribute__((ext_vector_type(8)))  float    v8f;

#define B_   2
#define S_   2048
#define DIM_ 1024
#define H_   16
#define DK_  64
#define BS_  (B_ * S_)

#define WMMA16(a, b, c) __builtin_amdgcn_wmma_f32_16x16x32_f16(false, (a), false, (b), (short)0, (c), false, false)

// ---------- WMMA fragment loaders (f32 source -> f16 operands) ----------

// A matrix 16x32 (MxK), row-major source with leading dim lda.
// Lanes 0-15: M=lane, K={0..7,16..23}; lanes 16-31: M=lane-16, K={8..15,24..31}
__device__ __forceinline__ v16h load_a_frag(const float* __restrict__ A, int lda,
                                            int row0, int k0, int lane) {
  int m = lane & 15;
  int half = lane >> 4;
  const float* p = A + (size_t)(row0 + m) * lda + (k0 + half * 8);
  float4 x0 = *(const float4*)(p + 0);
  float4 x1 = *(const float4*)(p + 4);
  float4 x2 = *(const float4*)(p + 16);
  float4 x3 = *(const float4*)(p + 20);
  v16h a;
  a[0]=(f16)x0.x;  a[1]=(f16)x0.y;  a[2]=(f16)x0.z;  a[3]=(f16)x0.w;
  a[4]=(f16)x1.x;  a[5]=(f16)x1.y;  a[6]=(f16)x1.z;  a[7]=(f16)x1.w;
  a[8]=(f16)x2.x;  a[9]=(f16)x2.y;  a[10]=(f16)x2.z; a[11]=(f16)x2.w;
  a[12]=(f16)x3.x; a[13]=(f16)x3.y; a[14]=(f16)x3.z; a[15]=(f16)x3.w;
  return a;
}

// Same A layout but from a generic/LDS pointer (scores strip), rows are local (0..15).
__device__ __forceinline__ v16h load_a_frag_smem(const float* Sm, int lda, int k0, int lane) {
  int m = lane & 15;
  int half = lane >> 4;
  const float* p = Sm + (size_t)m * lda + (k0 + half * 8);
  float4 x0 = *(const float4*)(p + 0);
  float4 x1 = *(const float4*)(p + 4);
  float4 x2 = *(const float4*)(p + 16);
  float4 x3 = *(const float4*)(p + 20);
  v16h a;
  a[0]=(f16)x0.x;  a[1]=(f16)x0.y;  a[2]=(f16)x0.z;  a[3]=(f16)x0.w;
  a[4]=(f16)x1.x;  a[5]=(f16)x1.y;  a[6]=(f16)x1.z;  a[7]=(f16)x1.w;
  a[8]=(f16)x2.x;  a[9]=(f16)x2.y;  a[10]=(f16)x2.z; a[11]=(f16)x2.w;
  a[12]=(f16)x3.x; a[13]=(f16)x3.y; a[14]=(f16)x3.z; a[15]=(f16)x3.w;
  return a;
}

// B matrix 32x16 (KxN) from row-major KxN source (column n strided by ldb).
__device__ __forceinline__ v16h load_b_strided(const float* __restrict__ Bm, int ldb,
                                               int k0, int col0, int lane) {
  int n = lane & 15;
  int kb = k0 + (lane >> 4) * 8;
  const float* p = Bm + (size_t)kb * ldb + (col0 + n);
  v16h b;
#pragma unroll
  for (int i = 0; i < 16; ++i) b[i] = (f16)p[(size_t)(i + (i & 8)) * ldb];
  return b;
}

// B matrix 32x16 (KxN) where B = T^T, T is NxK row-major: column n of B = row n of T (contiguous)
__device__ __forceinline__ v16h load_b_trans(const float* __restrict__ T, int ldt,
                                             int k0, int col0, int lane) {
  int n = lane & 15;
  int kb = k0 + (lane >> 4) * 8;
  const float* p = T + (size_t)(col0 + n) * ldt + kb;
  float4 x0 = *(const float4*)(p + 0);
  float4 x1 = *(const float4*)(p + 4);
  float4 x2 = *(const float4*)(p + 16);
  float4 x3 = *(const float4*)(p + 20);
  v16h b;
  b[0]=(f16)x0.x;  b[1]=(f16)x0.y;  b[2]=(f16)x0.z;  b[3]=(f16)x0.w;
  b[4]=(f16)x1.x;  b[5]=(f16)x1.y;  b[6]=(f16)x1.z;  b[7]=(f16)x1.w;
  b[8]=(f16)x2.x;  b[9]=(f16)x2.y;  b[10]=(f16)x2.z; b[11]=(f16)x2.w;
  b[12]=(f16)x3.x; b[13]=(f16)x3.y; b[14]=(f16)x3.z; b[15]=(f16)x3.w;
  return b;
}

// ---------- Kernel 1: per-head input projection; each wave computes a 16x64 strip (one head) ----------
__global__ void proj_kernel(const float* __restrict__ X,    // [BS, DIM]
                            const float* __restrict__ W,    // [H, DIM, DK]
                            const float* __restrict__ bias, // [H, DK]
                            float* __restrict__ Out) {      // [B*H, S, DK]
  int lane = threadIdx.x & 31;
  int t = blockIdx.x * 8 + (threadIdx.x >> 5);   // 256 row tiles x 16 heads = 4096 tasks
  int tm = t & 255;
  int h  = t >> 8;
  int row0 = tm * 16;
  const float* Wh = W + (size_t)h * DIM_ * DK_;
  v8f acc[4] = {};
#pragma unroll 2
  for (int k0 = 0; k0 < DIM_; k0 += 32) {
    v16h a = load_a_frag(X, DIM_, row0, k0, lane);
#pragma unroll
    for (int c = 0; c < 4; ++c) {
      v16h b = load_b_strided(Wh, DK_, k0, c * 16, lane);
      acc[c] = WMMA16(a, b, acc[c]);
    }
  }
  int colL = lane & 15;
  int rh = (lane >> 4) * 8;
#pragma unroll
  for (int c = 0; c < 4; ++c) {
    float bvv = bias[h * DK_ + c * 16 + colL];
#pragma unroll
    for (int r = 0; r < 8; ++r) {
      int row = row0 + rh + r;
      int bb = row >> 11;
      int s  = row & (S_ - 1);
      Out[(((size_t)bb * H_ + h) * S_ + s) * DK_ + c * 16 + colL] = acc[c][r] + bvv;
    }
  }
}

// ---------- Kernel 2: fused scores -> softmax -> attn.V per (b,h,16-row strip) ----------
// Block = 256 threads (8 waves). LDS: 16x2048 f32 score strip (128KB) + 8x256 split-K scratch.
__global__ void attn_fused_kernel(const float* __restrict__ Qh,  // [B*H, S, DK]
                                  const float* __restrict__ Kh,  // [B*H, S, DK]
                                  const float* __restrict__ Vh,  // [B*H, S, DK]
                                  float* __restrict__ Wt,        // [B*H, S, S]  (output weights)
                                  float* __restrict__ Abuf) {    // [B, S, DIM]  head-concat attn
  extern __shared__ float smem[];
  float* scores  = smem;             // 16 * 2048
  float* scratch = smem + 16 * S_;   // 8 * 256

  int lane = threadIdx.x & 31;
  int w = threadIdx.x >> 5;
  int bh = blockIdx.x >> 7;          // 128 row tiles per (b,h)
  int tm = blockIdx.x & 127;
  int row0 = tm * 16;
  int colL = lane & 15;
  int rh = (lane >> 4) * 8;

  const float* Q = Qh + (size_t)bh * S_ * DK_;
  const float* K = Kh + (size_t)bh * S_ * DK_;
  const float* V = Vh + (size_t)bh * S_ * DK_;

  // --- phase 1: score strip; Q fragments loaded once per wave, reused over 16 col tiles ---
  v16h a0 = load_a_frag(Q, DK_, row0, 0, lane);
  v16h a1 = load_a_frag(Q, DK_, row0, 32, lane);
  for (int tn = w * 16; tn < w * 16 + 16; ++tn) {
    int col0 = tn * 16;
    v8f acc = {};
    v16h b0 = load_b_trans(K, DK_, 0, col0, lane);
    acc = WMMA16(a0, b0, acc);
    v16h b1 = load_b_trans(K, DK_, 32, col0, lane);
    acc = WMMA16(a1, b1, acc);
#pragma unroll
    for (int r = 0; r < 8; ++r)
      scores[(size_t)(rh + r) * S_ + col0 + colL] = acc[r] * 0.125f;  // 1/sqrt(64)
  }
  __syncthreads();

  // --- phase 2: softmax rows 2w, 2w+1; write normalized weights to HBM non-temporally ---
  float* WtBase = Wt + ((size_t)bh * S_ + row0) * S_;
  for (int rr = 0; rr < 2; ++rr) {
    int row = w * 2 + rr;
    float* p = scores + (size_t)row * S_;
    float vals[64];
    float m = -3.4e38f;
#pragma unroll
    for (int i = 0; i < 64; ++i) { vals[i] = p[lane + i * 32]; m = fmaxf(m, vals[i]); }
#pragma unroll
    for (int off = 16; off > 0; off >>= 1) m = fmaxf(m, __shfl_xor(m, off, 32));
    float sum = 0.f;
#pragma unroll
    for (int i = 0; i < 64; ++i) { vals[i] = __expf(vals[i] - m); sum += vals[i]; }
#pragma unroll
    for (int off = 16; off > 0; off >>= 1) sum += __shfl_xor(sum, off, 32);
    float inv = 1.f / sum;
    float* g = WtBase + (size_t)row * S_;
#pragma unroll
    for (int i = 0; i < 64; ++i) {
      float x = vals[i] * inv;
      p[lane + i * 32] = x;
      __builtin_nontemporal_store(x, g + lane + i * 32);
    }
  }
  __syncthreads();

  // --- phase 3: attn = P(16x2048, LDS) @ V(2048x64); split-K over the 8 waves ---
  int c = w & 3;         // output col tile (16 cols of the 64)
  int khalf = w >> 2;    // K half: 0 -> [0,1024), 1 -> [1024,2048)
  v8f acc = {};
  int kbeg = khalf * 1024;
#pragma unroll 2
  for (int k0 = kbeg; k0 < kbeg + 1024; k0 += 32) {
    v16h a = load_a_frag_smem(scores, S_, k0, lane);
    v16h b = load_b_strided(V, DK_, k0, c * 16, lane);
    acc = WMMA16(a, b, acc);
  }
#pragma unroll
  for (int r = 0; r < 8; ++r) scratch[w * 256 + r * 32 + lane] = acc[r];
  __syncthreads();
  if (w < 4) {
    int bb = bh >> 4, h = bh & 15;
#pragma unroll
    for (int r = 0; r < 8; ++r) {
      float sumv = scratch[w * 256 + r * 32 + lane] + scratch[(w + 4) * 256 + r * 32 + lane];
      int s = row0 + rh + r;
      Abuf[((size_t)bb * S_ + s) * DIM_ + h * DK_ + c * 16 + colL] = sumv;
    }
  }
}

// ---------- Kernel 3: out = Abuf @ Wo + bo; each wave computes a 16x64 strip ----------
__global__ void outproj_kernel(const float* __restrict__ Abuf, // [BS, DIM]
                               const float* __restrict__ Wo,   // [DIM, DIM]
                               const float* __restrict__ bo,   // [DIM]
                               float* __restrict__ Out) {      // [BS, DIM]
  int lane = threadIdx.x & 31;
  int t = blockIdx.x * 8 + (threadIdx.x >> 5);   // 256 row tiles x 16 col groups
  int tm = t & 255;
  int cg = t >> 8;
  int row0 = tm * 16;
  int cbase = cg * 64;
  v8f acc[4] = {};
#pragma unroll 2
  for (int k0 = 0; k0 < DIM_; k0 += 32) {
    v16h a = load_a_frag(Abuf, DIM_, row0, k0, lane);
#pragma unroll
    for (int c = 0; c < 4; ++c) {
      v16h b = load_b_strided(Wo, DIM_, k0, cbase + c * 16, lane);
      acc[c] = WMMA16(a, b, acc[c]);
    }
  }
  int colL = lane & 15;
  int rh = (lane >> 4) * 8;
#pragma unroll
  for (int c = 0; c < 4; ++c) {
    float bvv = bo[cbase + c * 16 + colL];
#pragma unroll
    for (int r = 0; r < 8; ++r)
      Out[(size_t)(row0 + rh + r) * DIM_ + cbase + c * 16 + colL] = acc[c][r] + bvv;
  }
}

extern "C" void kernel_launch(void* const* d_in, const int* in_sizes, int n_in,
                              void* d_out, int out_size, void* d_ws, size_t ws_size,
                              hipStream_t stream) {
  const float* q  = (const float*)d_in[0];
  const float* k  = (const float*)d_in[1];
  const float* v  = (const float*)d_in[2];
  const float* Wq = (const float*)d_in[3];
  const float* bq = (const float*)d_in[4];
  const float* Wk = (const float*)d_in[5];
  const float* bk = (const float*)d_in[6];
  const float* Wv = (const float*)d_in[7];
  const float* bv = (const float*)d_in[8];
  const float* Wo = (const float*)d_in[9];
  const float* bo = (const float*)d_in[10];

  float* out = (float*)d_out;                       // [B, S, DIM]
  float* wts = out + (size_t)BS_ * DIM_;            // [B, H, S, S]

  const size_t headElems = (size_t)B_ * H_ * S_ * DK_;
  float* qh   = (float*)d_ws;
  float* kh   = qh + headElems;
  float* vh   = kh + headElems;
  float* abuf = vh + headElems;                     // [B, S, DIM]

  dim3 blk(256);
  // 4096 wave-tasks each, 8 waves/block
  proj_kernel<<<512, blk, 0, stream>>>(q, Wq, bq, qh);
  proj_kernel<<<512, blk, 0, stream>>>(k, Wk, bk, kh);
  proj_kernel<<<512, blk, 0, stream>>>(v, Wv, bv, vh);
  // one block per (b,h,16-row strip): 32 * 128 = 4096 blocks; 139,264 B dynamic LDS
  const size_t smemBytes = (size_t)(16 * S_ + 8 * 256) * sizeof(float);
  attn_fused_kernel<<<4096, blk, smemBytes, stream>>>(qh, kh, vh, wts, abuf);
  // 4096 wave-tasks
  outproj_kernel<<<512, blk, 0, stream>>>(abuf, Wo, bo, out);
}
